// MultiHeadSelfAttention_32916629356649
// MI455X (gfx1250) — matrix operand order repaired and checked by reading, not compiled, not run on hardware
//
#include <hip/hip_runtime.h>
#include <hip/hip_bf16.h>
#include <cstdint>

typedef __attribute__((ext_vector_type(16))) _Float16 v16h;
typedef __attribute__((ext_vector_type(8)))  _Float16 h8;
typedef __attribute__((ext_vector_type(8)))  float    v8f;

#define D_MODEL 1024
#define SEQ     2048
#define BATCH   4
#define NHEADS  16
#define HDIM    64

// ---------------------------------------------------------------------------
// CDNA5 async data movement: global -> LDS, tracked by ASYNCcnt (ISA ch.8/10).
// GV mode: vdst = LDS byte-offset VGPR, vaddr = 64-bit global address.
// ---------------------------------------------------------------------------
__device__ inline void async_copy_b128(const _Float16* g, _Float16* l) {
  uint32_t loff = (uint32_t)(uintptr_t)l;   // addr[31:0] = LDS offset
  asm volatile("global_load_async_to_lds_b128 %0, %1, off"
               :: "v"(loff), "v"(g) : "memory");
}
__device__ inline void wait_async0() {
  asm volatile("s_wait_asynccnt 0" ::: "memory");
}

// ---------------------------------------------------------------------------
// Fragment loaders (CDNA5 WMMA 16x16x32 f16 layouts, wave32)
// A 16x32 (MxK): lane L -> M=L&15; half hh=L>>4; elems 0..7 -> K=hh*8+i,
//                elems 8..15 -> K=16+hh*8+(i-8).  Two contiguous 8-half runs.
// B 32x16 (KxN): lane L -> N=L&15; same K order as A: two contiguous
//                8-half runs per lane (storage must be K-contiguous per N).
// C/D 16x16 f32: vgpr r + lane L -> (M = r + 8*hh, N = L&15).
// ---------------------------------------------------------------------------
__device__ inline v16h ldA_f16(const _Float16* p, int ld, int rowbase, int k0) {
  int lane = threadIdx.x & 31;
  int hh = lane >> 4, m = lane & 15;
  const _Float16* base = p + (size_t)(rowbase + m) * ld + k0;
  union { v16h v; h8 h[2]; } u;
  u.h[0] = *(const h8*)(base + hh * 8);
  u.h[1] = *(const h8*)(base + 16 + hh * 8);
  return u.v;
}

__device__ inline v16h ldA_f32cvt(const float* p, int ld, int rowbase, int k0) {
  int lane = threadIdx.x & 31;
  int hh = lane >> 4, m = lane & 15;
  const float* base = p + (size_t)(rowbase + m) * ld + k0;
  float4 a0 = *(const float4*)(base + hh * 8);
  float4 a1 = *(const float4*)(base + hh * 8 + 4);
  float4 b0 = *(const float4*)(base + 16 + hh * 8);
  float4 b1 = *(const float4*)(base + 16 + hh * 8 + 4);
  v16h r;
  r[0]=(_Float16)a0.x; r[1]=(_Float16)a0.y; r[2]=(_Float16)a0.z; r[3]=(_Float16)a0.w;
  r[4]=(_Float16)a1.x; r[5]=(_Float16)a1.y; r[6]=(_Float16)a1.z; r[7]=(_Float16)a1.w;
  r[8]=(_Float16)b0.x; r[9]=(_Float16)b0.y; r[10]=(_Float16)b0.z; r[11]=(_Float16)b0.w;
  r[12]=(_Float16)b1.x; r[13]=(_Float16)b1.y; r[14]=(_Float16)b1.z; r[15]=(_Float16)b1.w;
  return r;
}

__device__ inline v16h ldB_f16(const _Float16* p, int ld, int nbase, int k0) {
  int lane = threadIdx.x & 31;
  int hh = lane >> 4, n = lane & 15;
  const _Float16* base = p + (size_t)(nbase + n) * ld + k0;
  union { v16h v; h8 h[2]; } u;
  u.h[0] = *(const h8*)(base + hh * 8);
  u.h[1] = *(const h8*)(base + 16 + hh * 8);
  return u.v;
}

__device__ inline v8f wmma16x16x32(v16h a, v16h b, v8f c) {
  return __builtin_amdgcn_wmma_f32_16x16x32_f16(false, a, false, b,
                                                (short)0, c, false, false);
}

// ---------------------------------------------------------------------------
// Kernel 0: transpose+convert the four weight matrices to f16, Wt[n][k]=W[k][n]
// ---------------------------------------------------------------------------
__global__ __launch_bounds__(256)
void prep_weights_kernel(const float* Wq, const float* Wk, const float* Wv,
                         const float* Wo, _Float16* WtQ, _Float16* WtK,
                         _Float16* WtV, _Float16* WtO) {
  const float* src; _Float16* dst;
  switch (blockIdx.y) {
    case 0: src = Wq; dst = WtQ; break;
    case 1: src = Wk; dst = WtK; break;
    case 2: src = Wv; dst = WtV; break;
    default: src = Wo; dst = WtO; break;
  }
  int idx = blockIdx.x * 256 + threadIdx.x;           // 0 .. 1M-1
  int k = idx >> 10, n = idx & 1023;
  dst[(size_t)n * D_MODEL + k] = (_Float16)src[idx];
}

// ---------------------------------------------------------------------------
// Kernel 1: fused Q/K/V projection. blockIdx.z selects matrix.
// 8 waves; wave computes 16 rows x 64 cols. Weight tile (64 n x 32 k, 4 KB)
// is async-staged into LDS once per block, double-buffered.
// ---------------------------------------------------------------------------
__global__ __launch_bounds__(256)
void qkv_proj_kernel(const float* __restrict__ x,
                     const _Float16* __restrict__ WtQ,
                     const _Float16* __restrict__ WtK,
                     const _Float16* __restrict__ WtV,
                     const float* __restrict__ bq,
                     const float* __restrict__ bk,
                     const float* __restrict__ bv,
                     _Float16* __restrict__ Qh,
                     _Float16* __restrict__ Kh,
                     _Float16* __restrict__ Vt) {
  __shared__ __align__(16) _Float16 wbuf[2][64 * 32];

  int z = blockIdx.z;
  const _Float16* Wt = (z == 0) ? WtQ : (z == 1) ? WtK : WtV;
  const float* bias  = (z == 0) ? bq  : (z == 1) ? bk  : bv;

  int t = threadIdx.x;
  int w = t >> 5, lane = t & 31;
  int hh = lane >> 4, nl = lane & 15;
  int mbase = blockIdx.y * 128 + w * 16;
  int nbase = blockIdx.x * 64;

  // one b128 per thread stages the 64x32 f16 weight tile
  int wrow = t >> 2, wch = t & 3;
  auto stageW = [&](int k0, int bi) {
    async_copy_b128(Wt + (size_t)(nbase + wrow) * D_MODEL + k0 + wch * 8,
                    &wbuf[bi][wrow * 32 + wch * 8]);
  };

  v8f acc[4] = {};
  stageW(0, 0);
  wait_async0();
  __syncthreads();

  for (int k0 = 0; k0 < D_MODEL; k0 += 32) {
    int bi = (k0 >> 5) & 1;
    if (k0 + 32 < D_MODEL) stageW(k0 + 32, bi ^ 1);

    // issue all fragment loads first, then the WMMA group
    v16h a  = ldA_f32cvt(x, D_MODEL, mbase, k0);
    v16h b0 = ldB_f16(wbuf[bi], 32, 0,  0);
    v16h b1 = ldB_f16(wbuf[bi], 32, 16, 0);
    v16h b2 = ldB_f16(wbuf[bi], 32, 32, 0);
    v16h b3 = ldB_f16(wbuf[bi], 32, 48, 0);
    acc[0] = wmma16x16x32(a, b0, acc[0]);
    acc[1] = wmma16x16x32(a, b1, acc[1]);
    acc[2] = wmma16x16x32(a, b2, acc[2]);
    acc[3] = wmma16x16x32(a, b3, acc[3]);

    wait_async0();
    __syncthreads();
  }

#pragma unroll
  for (int j = 0; j < 4; ++j) {
    int n = nbase + 16 * j + nl;
    float bb = bias[n];
    int h = n >> 6, d = n & 63;
#pragma unroll
    for (int r = 0; r < 8; ++r) {
      int mg = mbase + r + 8 * hh;           // global token row
      int b  = mg >> 11, s = mg & 2047;
      float val = acc[j][r] + bb;
      size_t bh = (size_t)(b * NHEADS + h);
      if (z == 0)      Qh[(bh * SEQ + s) * HDIM + d] = (_Float16)val;
      else if (z == 1) Kh[(bh * SEQ + s) * HDIM + d] = (_Float16)val;
      else             Vt[(bh * HDIM + d) * SEQ + s] = (_Float16)val;
    }
  }
}

// ---------------------------------------------------------------------------
// Kernel 2: flash attention. 8 waves; wave owns 16 query rows.
// K tile (32 keys x 64, 4 KB) and V tile (64 feat x 32 keys, 4 KB) are
// async-staged into LDS per 32-key step, double-buffered to overlap with
// the 8 WMMAs + online softmax of the current step.
// ---------------------------------------------------------------------------
#define PL_LD 40   // halves per LDS P row (16B-aligned runs, conflict padding)

__global__ __launch_bounds__(256)
void attn_kernel(const _Float16* __restrict__ Qh,
                 const _Float16* __restrict__ Kh,
                 const _Float16* __restrict__ Vt,
                 _Float16* __restrict__ Oh) {
  __shared__ __align__(16) _Float16 kbuf[2][32 * 64];
  __shared__ __align__(16) _Float16 vbuf[2][64 * 32];
  __shared__ __align__(16) _Float16 plds[8 * 16 * PL_LD];

  int t = threadIdx.x;
  int w = t >> 5, lane = t & 31;
  int hh = lane >> 4, nl = lane & 15;
  int b = blockIdx.z, h = blockIdx.y;
  int q0 = blockIdx.x * 128 + w * 16;

  size_t bh = (size_t)(b * NHEADS + h);
  const _Float16* Qp = Qh + bh * SEQ * HDIM;
  const _Float16* Kp = Kh + bh * SEQ * HDIM;
  const _Float16* Vp = Vt + bh * HDIM * SEQ;
  _Float16* pl = plds + w * 16 * PL_LD;

  // cooperative tile staging: one b128 per thread per tile
  int krow = t >> 3, kch = t & 7;   // 32 rows x 8 chunks (K tile)
  int vrow = t >> 2, vch = t & 3;   // 64 rows x 4 chunks (V tile)
  auto stageKV = [&](int k0, int bi) {
    async_copy_b128(Kp + (size_t)(k0 + krow) * HDIM + kch * 8,
                    &kbuf[bi][krow * 64 + kch * 8]);
    async_copy_b128(Vp + (size_t)vrow * SEQ + k0 + vch * 8,
                    &vbuf[bi][vrow * 32 + vch * 8]);
  };

  v16h qf0 = ldA_f16(Qp, HDIM, q0, 0);
  v16h qf1 = ldA_f16(Qp, HDIM, q0, 32);

  v8f o[4] = {};
  float mst[8], lst[8];
#pragma unroll
  for (int r = 0; r < 8; ++r) { mst[r] = -1e30f; lst[r] = 0.0f; }

  const float scale = 0.125f;   // 1/sqrt(64)

  stageKV(0, 0);
  wait_async0();
  __syncthreads();

  for (int k0 = 0; k0 < SEQ; k0 += 32) {
    int bi = (k0 >> 5) & 1;
    if (k0 + 32 < SEQ) stageKV(k0 + 32, bi ^ 1);

    // scores: 16 q-rows x 32 keys, Dh=64 in two K-chunks (from LDS K tile).
    // Load all four K fragments first so ds_loads clause and overlap WMMA.
    v16h kf00 = ldB_f16(kbuf[bi], 64, 0,  0);
    v16h kf01 = ldB_f16(kbuf[bi], 64, 0,  32);
    v16h kf10 = ldB_f16(kbuf[bi], 64, 16, 0);
    v16h kf11 = ldB_f16(kbuf[bi], 64, 16, 32);
    v8f c0 = {}, c1 = {};
    c0 = wmma16x16x32(qf0, kf00, c0);
    c0 = wmma16x16x32(qf1, kf01, c0);
    c1 = wmma16x16x32(qf0, kf10, c1);
    c1 = wmma16x16x32(qf1, kf11, c1);

    // online softmax; row m = r + 8*hh spans the 16 lanes of this half-wave
#pragma unroll
    for (int r = 0; r < 8; ++r) {
      float s0 = c0[r] * scale, s1 = c1[r] * scale;
      float mx = fmaxf(s0, s1);
#pragma unroll
      for (int off = 1; off < 16; off <<= 1)
        mx = fmaxf(mx, __shfl_xor(mx, off, 32));
      float nm = fmaxf(mst[r], mx);
      float al = __expf(mst[r] - nm);
      float p0 = __expf(s0 - nm);
      float p1 = __expf(s1 - nm);
      float rs = p0 + p1;
#pragma unroll
      for (int off = 1; off < 16; off <<= 1)
        rs += __shfl_xor(rs, off, 32);
      lst[r] = lst[r] * al + rs;
      mst[r] = nm;
      o[0][r] *= al; o[1][r] *= al; o[2][r] *= al; o[3][r] *= al;
      int m = r + 8 * hh;
      pl[m * PL_LD + nl]      = (_Float16)p0;
      pl[m * PL_LD + nl + 16] = (_Float16)p1;
    }

    // P (16x32 f16) back from LDS in A layout; accumulate P @ V (V from LDS)
    v16h pf  = ldA_f16(pl, PL_LD, 0, 0);
    v16h vf0 = ldB_f16(vbuf[bi], 32, 0,  0);
    v16h vf1 = ldB_f16(vbuf[bi], 32, 16, 0);
    v16h vf2 = ldB_f16(vbuf[bi], 32, 32, 0);
    v16h vf3 = ldB_f16(vbuf[bi], 32, 48, 0);
    o[0] = wmma16x16x32(pf, vf0, o[0]);
    o[1] = wmma16x16x32(pf, vf1, o[1]);
    o[2] = wmma16x16x32(pf, vf2, o[2]);
    o[3] = wmma16x16x32(pf, vf3, o[3]);

    wait_async0();     // next K/V tile resident
    __syncthreads();   // everyone done reading current tiles
  }

  // normalize and store attention output [B,S,D] f16
#pragma unroll
  for (int r = 0; r < 8; ++r) {
    float inv = 1.0f / lst[r];
    int row = b * SEQ + q0 + r + 8 * hh;
#pragma unroll
    for (int j = 0; j < 4; ++j) {
      int col = h * HDIM + 16 * j + nl;
      Oh[(size_t)row * D_MODEL + col] = (_Float16)(o[j][r] * inv);
    }
  }
}

// ---------------------------------------------------------------------------
// Kernel 3: output projection  out = Oh @ Wo + bo   (f16 GEMM, f32 out)
// Weight tile async-staged into LDS, double-buffered.
// ---------------------------------------------------------------------------
__global__ __launch_bounds__(256)
void oproj_kernel(const _Float16* __restrict__ Oh,
                  const _Float16* __restrict__ WtO,
                  const float* __restrict__ bo,
                  float* __restrict__ out) {
  __shared__ __align__(16) _Float16 wbuf[2][64 * 32];

  int t = threadIdx.x;
  int w = t >> 5, lane = t & 31;
  int hh = lane >> 4, nl = lane & 15;
  int mbase = blockIdx.y * 128 + w * 16;
  int nbase = blockIdx.x * 64;

  int wrow = t >> 2, wch = t & 3;
  auto stageW = [&](int k0, int bi) {
    async_copy_b128(WtO + (size_t)(nbase + wrow) * D_MODEL + k0 + wch * 8,
                    &wbuf[bi][wrow * 32 + wch * 8]);
  };

  v8f acc[4] = {};
  stageW(0, 0);
  wait_async0();
  __syncthreads();

  for (int k0 = 0; k0 < D_MODEL; k0 += 32) {
    int bi = (k0 >> 5) & 1;
    if (k0 + 32 < D_MODEL) stageW(k0 + 32, bi ^ 1);

    v16h a  = ldA_f16(Oh, D_MODEL, mbase, k0);
    v16h b0 = ldB_f16(wbuf[bi], 32, 0,  0);
    v16h b1 = ldB_f16(wbuf[bi], 32, 16, 0);
    v16h b2 = ldB_f16(wbuf[bi], 32, 32, 0);
    v16h b3 = ldB_f16(wbuf[bi], 32, 48, 0);
    acc[0] = wmma16x16x32(a, b0, acc[0]);
    acc[1] = wmma16x16x32(a, b1, acc[1]);
    acc[2] = wmma16x16x32(a, b2, acc[2]);
    acc[3] = wmma16x16x32(a, b3, acc[3]);

    wait_async0();
    __syncthreads();
  }

#pragma unroll
  for (int j = 0; j < 4; ++j) {
    int n = nbase + 16 * j + nl;
    float bb = bo[n];
#pragma unroll
    for (int r = 0; r < 8; ++r)
      out[(size_t)(mbase + r + 8 * hh) * D_MODEL + n] = acc[j][r] + bb;
  }
}

// ---------------------------------------------------------------------------
extern "C" void kernel_launch(void* const* d_in, const int* in_sizes, int n_in,
                              void* d_out, int out_size, void* d_ws, size_t ws_size,
                              hipStream_t stream) {
  const float* x  = (const float*)d_in[0];
  const float* Wq = (const float*)d_in[1];
  const float* bq = (const float*)d_in[2];
  const float* Wk = (const float*)d_in[3];
  const float* bk = (const float*)d_in[4];
  const float* Wv = (const float*)d_in[5];
  const float* bv = (const float*)d_in[6];
  const float* Wo = (const float*)d_in[7];
  const float* bo = (const float*)d_in[8];
  float* out = (float*)d_out;

  const size_t WELEM = (size_t)D_MODEL * D_MODEL;                 // 1M
  const size_t QELEM = (size_t)BATCH * NHEADS * SEQ * HDIM;       // 8M

  _Float16* WtQ = (_Float16*)d_ws;
  _Float16* WtK = WtQ + WELEM;
  _Float16* WtV = WtK + WELEM;
  _Float16* WtO = WtV + WELEM;
  _Float16* Qh  = WtO + WELEM;
  _Float16* Kh  = Qh + QELEM;
  _Float16* Vt  = Kh + QELEM;
  _Float16* Oh  = Vt + QELEM;   // [B*S, D] f16

  prep_weights_kernel<<<dim3(WELEM / 256, 4), 256, 0, stream>>>(
      Wq, Wk, Wv, Wo, WtQ, WtK, WtV, WtO);

  qkv_proj_kernel<<<dim3(D_MODEL / 64, (BATCH * SEQ) / 128, 3), 256, 0, stream>>>(
      x, WtQ, WtK, WtV, bq, bk, bv, Qh, Kh, Vt);

  attn_kernel<<<dim3(SEQ / 128, NHEADS, BATCH), 256, 0, stream>>>(
      Qh, Kh, Vt, Oh);

  oproj_kernel<<<dim3(D_MODEL / 64, (BATCH * SEQ) / 128), 256, 0, stream>>>(
      Oh, WtO, bo, out);
}
